// QLSTM_65481071408256
// MI455X (gfx1250) — compile-verified, matrix operand order checked by reading, not run on hardware
//
#include <hip/hip_runtime.h>

typedef __attribute__((ext_vector_type(16))) _Float16  v16h;
typedef __attribute__((ext_vector_type(8)))  float     v8f;
typedef __attribute__((ext_vector_type(4)))  unsigned  v4u;

namespace {

constexpr int TSTEPS = 512;
constexpr int BATCH  = 1024;
constexpr int INDIM  = 64;
constexpr int HID    = 8;
constexpr int CDIM   = 72;   // INPUT_DIM + HIDDEN
constexpr int KPAD   = 96;   // K padded to 3 x 32 for f16 WMMA
constexpr int BPW    = 4;    // batch rows per workgroup

// WMMA f16 fragment: per ISA 7.12.2, lane holds two contiguous 8-half runs.
union FragU {
  v16h h;
  v4u  u4[2];
};

// ---------------------------------------------------------------------------
// Quantum state: 256 complex amps per circuit, one circuit per wave32.
// Amp index k = (lane << 3) | r ;  bits 0..2 of k live in registers (r),
// bits 3..7 live across lanes.  RX on a register bit is pure VALU; RX on a
// lane bit is a shfl_xor exchange (RX matrix [[c,-is],[-is,c]] is symmetric,
// so new = c*a - i*s*partner uniformly on both sides of the pair).
// ---------------------------------------------------------------------------

template<int B>
__device__ __forceinline__ void rx_apply(float (&ar)[8], float (&ai)[8],
                                         float c, float s) {
  if constexpr (B >= 3) {
    constexpr int m = 1 << (B - 3);
#pragma unroll
    for (int r = 0; r < 8; ++r) {
      float pr = __shfl_xor(ar[r], m, 32);
      float pi = __shfl_xor(ai[r], m, 32);
      float nr = c * ar[r] + s * pi;
      float ni = c * ai[r] - s * pr;
      ar[r] = nr; ai[r] = ni;
    }
  } else {
    constexpr int mb = 1 << B;
#pragma unroll
    for (int r0 = 0; r0 < 8; ++r0) {
      if ((r0 & mb) == 0) {
        const int r1 = r0 | mb;
        float a0r = ar[r0], a0i = ai[r0], a1r = ar[r1], a1i = ai[r1];
        ar[r0] = c * a0r + s * a1i;  ai[r0] = c * a0i - s * a1r;
        ar[r1] = c * a1r + s * a0i;  ai[r1] = c * a1i - s * a0r;
      }
    }
  }
}

// CNOT with control bit BC and target bit BT of the amp index.
template<int BC, int BT>
__device__ __forceinline__ void cnot_apply(float (&ar)[8], float (&ai)[8],
                                           int lane) {
  if constexpr (BC >= 3 && BT >= 3) {
    constexpr int tm = 1 << (BT - 3);
    const bool cb = (lane >> (BC - 3)) & 1;
#pragma unroll
    for (int r = 0; r < 8; ++r) {
      float er = __shfl_xor(ar[r], tm, 32);
      float ei = __shfl_xor(ai[r], tm, 32);
      ar[r] = cb ? er : ar[r];
      ai[r] = cb ? ei : ai[r];
    }
  } else if constexpr (BC >= 3 && BT < 3) {
    const bool cb = (lane >> (BC - 3)) & 1;
    constexpr int tb = 1 << BT;
#pragma unroll
    for (int r0 = 0; r0 < 8; ++r0) {
      if ((r0 & tb) == 0) {
        const int r1 = r0 | tb;
        float t0r = ar[r0], t0i = ai[r0];
        ar[r0] = cb ? ar[r1] : ar[r0];  ai[r0] = cb ? ai[r1] : ai[r0];
        ar[r1] = cb ? t0r    : ar[r1];  ai[r1] = cb ? t0i    : ai[r1];
      }
    }
  } else if constexpr (BC < 3 && BT >= 3) {
    constexpr int tm  = 1 << (BT - 3);
    constexpr int cbm = 1 << BC;
#pragma unroll
    for (int r = 0; r < 8; ++r) {
      if (r & cbm) {  // unconditional lane exchange for these registers
        ar[r] = __shfl_xor(ar[r], tm, 32);
        ai[r] = __shfl_xor(ai[r], tm, 32);
      }
    }
  } else {
    constexpr int cbm = 1 << BC;
    constexpr int tb  = 1 << BT;
#pragma unroll
    for (int r = 0; r < 8; ++r) {
      if ((r & cbm) && !(r & tb)) {
        const int r1 = r | tb;
        float tr = ar[r]; ar[r] = ar[r1]; ar[r1] = tr;
        float ti = ai[r]; ai[r] = ai[r1]; ai[r1] = ti;
      }
    }
  }
}

} // namespace

__global__ __launch_bounds__(256)
void qlstm_kernel(const float* __restrict__ x,
                  const float* __restrict__ Wf, const float* __restrict__ bf,
                  const float* __restrict__ Wi, const float* __restrict__ bi,
                  const float* __restrict__ Wu, const float* __restrict__ bu,
                  const float* __restrict__ Wo, const float* __restrict__ bo,
                  const float* __restrict__ Tf, const float* __restrict__ Ti,
                  const float* __restrict__ Tu, const float* __restrict__ To,
                  float* __restrict__ out)
{
  // N-major weight matrix: row n = gate*8+wire, K contiguous -> fragment
  // loads are two ds_load_b128 per lane.
  __shared__ __align__(16) _Float16 BmatT[32][KPAD];
  __shared__ __align__(16) _Float16 Amat[16][KPAD];   // comb rows (4 used)
  __shared__ float csT[4][2][8][2];                   // cos/sin half-thetas
  __shared__ float biasL[32];
  __shared__ float angL[BPW][32];                     // encoder angles
  __shared__ float gateL[BPW][4][8];                  // activated expectations
  __shared__ float hL[BPW][8];
  __shared__ float cL[BPW][8];

  const int tid    = threadIdx.x;
  const int lane   = tid & 31;
  const int wv     = tid >> 5;
  const int batch0 = blockIdx.x * BPW;

  // ---- one-time setup: weights -> f16 B-matrix (transposed), thetas ----
  for (int idx = tid; idx < 32 * KPAD; idx += 256) {
    const int n = idx / KPAD, k = idx % KPAD;
    const int g = n >> 3,     w = n & 7;
    float v = 0.f;
    if (k < CDIM) {
      const float* W = (g == 0) ? Wf : (g == 1) ? Wi : (g == 2) ? Wu : Wo;
      v = W[w * CDIM + k];
    }
    BmatT[n][k] = (_Float16)v;
  }
  for (int idx = tid; idx < 16 * KPAD; idx += 256)
    (&Amat[0][0])[idx] = (_Float16)0.f;
  if (tid < 64) {
    const int g = tid >> 4, d = (tid >> 3) & 1, w = tid & 7;
    const float* Tt = (g == 0) ? Tf : (g == 1) ? Ti : (g == 2) ? Tu : To;
    const float th = 0.5f * Tt[d * 8 + w];
    csT[g][d][w][0] = __cosf(th);
    csT[g][d][w][1] = __sinf(th);
  }
  if (tid < 32) {
    const int g = tid >> 3, w = tid & 7;
    const float* bb = (g == 0) ? bf : (g == 1) ? bi : (g == 2) ? bu : bo;
    biasL[tid] = bb[w];
    (&hL[0][0])[tid] = 0.f;
    (&cL[0][0])[tid] = 0.f;
  }
  __syncthreads();

  // ---- hoist the time-invariant B fragments into wave-0 registers ----
  // Fragment layout (16-bit, 16x32): lane l -> N = l&15, runs of K at
  // grp*8 + kc*32 + {0..7} and +16.  48 VGPRs total for all 6 fragments.
  FragU bfrag[3][2];
  if (tid < 32) {
    const int Mr = lane & 15, grp = lane >> 4;
#pragma unroll
    for (int kc = 0; kc < 3; ++kc) {
#pragma unroll
      for (int tN = 0; tN < 2; ++tN) {
        const _Float16* bp = &BmatT[tN * 16 + Mr][kc * 32 + grp * 8];
        bfrag[kc][tN].u4[0] = *(const v4u*)bp;
        bfrag[kc][tN].u4[1] = *(const v4u*)(bp + 16);
      }
    }
  }

  for (int t = 0; t < TSTEPS; ++t) {
    // ---- stage comb = [x_t , h] into the A-matrix (f16) ----
    {
      const int row = tid >> 6, col = tid & 63;
      const float* xp = &x[((size_t)t * BATCH + batch0 + row) * INDIM + col];
      Amat[row][col] = (_Float16)(*xp);
      if (t + 1 < TSTEPS)
        __builtin_prefetch(xp + (size_t)BATCH * INDIM, 0, 1);
      if (tid < 32) {
        const int r2 = tid >> 3, w2 = tid & 7;
        Amat[r2][INDIM + w2] = (_Float16)hL[r2][w2];
      }
    }
    __syncthreads();

    // ---- wave 0: angles[16,32] = A[16,96] x B[96,32] via WMMA f16 ----
    if (tid < 32) {
      const int Mr = lane & 15, grp = lane >> 4;
      v8f acc0 = {}; v8f acc1 = {};
#pragma unroll
      for (int kc = 0; kc < 3; ++kc) {
        FragU af;
        const _Float16* ap = &Amat[Mr][kc * 32 + grp * 8];
        af.u4[0] = *(const v4u*)ap;          // K = grp*8 + 0..7
        af.u4[1] = *(const v4u*)(ap + 16);   // K = grp*8 + 16..23
        acc0 = __builtin_amdgcn_wmma_f32_16x16x32_f16(
            false, af.h, false, bfrag[kc][0].h, (short)0, acc0, false, false);
        acc1 = __builtin_amdgcn_wmma_f32_16x16x32_f16(
            false, af.h, false, bfrag[kc][1].h, (short)0, acc1, false, false);
      }
      if (lane < 16) {
#pragma unroll
        for (int bi_ = 0; bi_ < BPW; ++bi_) {   // D: VGPR j <-> M=j, lanes0-15
          angL[bi_][lane]      = acc0[bi_] + biasL[lane];
          angL[bi_][16 + lane] = acc1[bi_] + biasL[16 + lane];
        }
      }
    }
    __syncthreads();

    // ---- 16 circuits per block: 2 per wave ----
    for (int j = 0; j < 2; ++j) {
      const int cid = wv * 2 + j;
      const int bl  = cid >> 2;   // batch row in block
      const int g   = cid & 3;    // gate (0=f,1=i,2=u,3=o)

      // encoder: product state |psi> = prod RX(angle_w)|0>
      // bit b of k is acted on by wire 7-b
      float cb[8], sb[8];
#pragma unroll
      for (int b = 0; b < 8; ++b) {
        const float h = 0.5f * angL[bl][g * 8 + (7 - b)];
        cb[b] = __cosf(h);
        sb[b] = __sinf(h);
      }
      float mlane = 1.f;
#pragma unroll
      for (int b = 3; b < 8; ++b)
        mlane *= ((lane >> (b - 3)) & 1) ? sb[b] : cb[b];
      const int pcl = __popc(lane);

      float ar[8], ai[8];
#pragma unroll
      for (int r = 0; r < 8; ++r) {
        float m = mlane;
        m *= (r & 1) ? sb[0] : cb[0];
        m *= (r & 2) ? sb[1] : cb[1];
        m *= (r & 4) ? sb[2] : cb[2];
        const int p = (pcl + __popc(r)) & 3;   // amp = m * (-i)^p
        ar[r] = (p == 0) ? m : ((p == 2) ? -m : 0.f);
        ai[r] = (p == 3) ? m : ((p == 1) ? -m : 0.f);
      }

      // trainable layers: wires 0..7 -> bits 7..0 ; ring CNOTs
      for (int d = 0; d < 2; ++d) {
        rx_apply<7>(ar, ai, csT[g][d][0][0], csT[g][d][0][1]);
        rx_apply<6>(ar, ai, csT[g][d][1][0], csT[g][d][1][1]);
        rx_apply<5>(ar, ai, csT[g][d][2][0], csT[g][d][2][1]);
        rx_apply<4>(ar, ai, csT[g][d][3][0], csT[g][d][3][1]);
        rx_apply<3>(ar, ai, csT[g][d][4][0], csT[g][d][4][1]);
        rx_apply<2>(ar, ai, csT[g][d][5][0], csT[g][d][5][1]);
        rx_apply<1>(ar, ai, csT[g][d][6][0], csT[g][d][6][1]);
        rx_apply<0>(ar, ai, csT[g][d][7][0], csT[g][d][7][1]);
        cnot_apply<7, 6>(ar, ai, lane);   // wire0->1
        cnot_apply<6, 5>(ar, ai, lane);
        cnot_apply<5, 4>(ar, ai, lane);
        cnot_apply<4, 3>(ar, ai, lane);
        cnot_apply<3, 2>(ar, ai, lane);
        cnot_apply<2, 1>(ar, ai, lane);
        cnot_apply<1, 0>(ar, ai, lane);
        cnot_apply<0, 7>(ar, ai, lane);   // wire7->0
      }

      // Pauli-Z expectations: e[b] = sum_k (1-2*bit_b(k)) |amp_k|^2
      float S = 0.f, S0 = 0.f, S1 = 0.f, S2 = 0.f;
#pragma unroll
      for (int r = 0; r < 8; ++r) {
        const float p = ar[r] * ar[r] + ai[r] * ai[r];
        S += p;
        if (r & 1) S0 += p;
        if (r & 2) S1 += p;
        if (r & 4) S2 += p;
      }
      float e[8];
      e[0] = S - 2.f * S0;
      e[1] = S - 2.f * S1;
      e[2] = S - 2.f * S2;
#pragma unroll
      for (int b = 3; b < 8; ++b)
        e[b] = ((lane >> (b - 3)) & 1) ? -S : S;
#pragma unroll
      for (int m = 1; m < 32; m <<= 1) {
#pragma unroll
        for (int b = 0; b < 8; ++b)
          e[b] += __shfl_xor(e[b], m, 32);
      }

      if (lane == 0) {
#pragma unroll
        for (int b = 0; b < 8; ++b) {   // wire w = 7 - b
          const float v = e[b];
          const float act = (g == 2) ? tanhf(v)
                                     : 1.f / (1.f + __expf(-v));
          gateL[bl][g][7 - b] = act;
        }
      }
    }
    __syncthreads();

    // ---- LSTM cell update + emit h_t ----
    if (tid < 32) {
      const int b2 = tid >> 3, w2 = tid & 7;
      const float f  = gateL[b2][0][w2];
      const float ig = gateL[b2][1][w2];
      const float gg = gateL[b2][2][w2];
      const float og = gateL[b2][3][w2];
      const float cn = f * cL[b2][w2] + ig * gg;
      const float hn = og * tanhf(cn);
      cL[b2][w2] = cn;
      hL[b2][w2] = hn;
      out[((size_t)t * BATCH + batch0 + b2) * HID + w2] = hn;
    }
    __syncthreads();
  }

  // ---- final hx, cx ----
  if (tid < 32) {
    const int b2 = tid >> 3, w2 = tid & 7;
    const size_t base = (size_t)TSTEPS * BATCH * HID;
    out[base + (size_t)(batch0 + b2) * HID + w2] = hL[b2][w2];
    out[base + (size_t)BATCH * HID + (size_t)(batch0 + b2) * HID + w2] =
        cL[b2][w2];
  }
}

extern "C" void kernel_launch(void* const* d_in, const int* in_sizes, int n_in,
                              void* d_out, int out_size, void* d_ws,
                              size_t ws_size, hipStream_t stream) {
  (void)in_sizes; (void)n_in; (void)out_size; (void)d_ws; (void)ws_size;
  const float* x  = (const float*)d_in[0];
  const float* Wf = (const float*)d_in[1];
  const float* bf = (const float*)d_in[2];
  const float* Wi = (const float*)d_in[3];
  const float* bi = (const float*)d_in[4];
  const float* Wu = (const float*)d_in[5];
  const float* bu = (const float*)d_in[6];
  const float* Wo = (const float*)d_in[7];
  const float* bo = (const float*)d_in[8];
  const float* Tf = (const float*)d_in[9];
  const float* Ti = (const float*)d_in[10];
  const float* Tu = (const float*)d_in[11];
  const float* To = (const float*)d_in[12];
  float* out = (float*)d_out;

  dim3 grid(BATCH / BPW);   // 256 blocks, each owns 4 batch rows for all T
  dim3 block(256);          // 8 waves: 1 circuit per wave, 2 rounds per step
  qlstm_kernel<<<grid, block, 0, stream>>>(x, Wf, bf, Wi, bi, Wu, bu, Wo, bo,
                                           Tf, Ti, Tu, To, out);
}
